// MambaLayer_34952443855213
// MI455X (gfx1250) — matrix operand order repaired and checked by reading, not compiled, not run on hardware
//
#include <hip/hip_runtime.h>
#include <hip/hip_bf16.h>

// ---------------------------------------------------------------------------
// Mamba layer forward for MI455X (gfx1250).
// bf16 WMMA GEMMs with async global->LDS double-buffered staging + register scan.
// ---------------------------------------------------------------------------

typedef unsigned short u16;
typedef __attribute__((ext_vector_type(16))) __bf16 v16bf;
typedef __attribute__((ext_vector_type(8)))  float  v8f;

struct Frag32 { uint4 lo; uint4 hi; };

__device__ __forceinline__ v16bf make_frag(uint4 lo, uint4 hi) {
  Frag32 f{lo, hi};
  return __builtin_bit_cast(v16bf, f);
}

__device__ __forceinline__ u16 f2bf(float x) {
  unsigned u = __builtin_bit_cast(unsigned, x);
  u += 0x7FFFu + ((u >> 16) & 1u);            // round-to-nearest-even
  return (u16)(u >> 16);
}

__device__ __forceinline__ float siluf(float x) {
  return x / (1.f + __expf(-x));
}

// 32-bit LDS byte offset of a pointer into a __shared__ array.
__device__ __forceinline__ unsigned lds_off(const void* p) {
  return (unsigned)(unsigned long long)(const __attribute__((address_space(3))) void*)p;
}

// Async DMA: 16 bytes per enabled lane, global -> LDS. Tracked by ASYNCcnt.
__device__ __forceinline__ void async_copy_b128(unsigned lds_byte_off, const void* gaddr) {
  asm volatile("global_load_async_to_lds_b128 %0, %1, off"
               :: "v"(lds_byte_off), "v"(gaddr)
               : "memory");
}

__device__ __forceinline__ void wait_async0() {
  asm volatile("s_wait_asynccnt 0x0" ::: "memory");
}

// ---------------------------------------------------------------------------
// Small elementwise kernels: f32->bf16 conversions / packing
// ---------------------------------------------------------------------------
__global__ void cvt_bf16(const float* __restrict__ in, u16* __restrict__ out, int n) {
  for (int i = blockIdx.x * blockDim.x + threadIdx.x; i < n; i += gridDim.x * blockDim.x)
    out[i] = f2bf(in[i]);
}

// conv_w (O=2048, I=2048, Kc=4) -> packed bf16 (O, Kc*2048) with K-major segments:
// out[o*8192 + seg*2048 + i] = conv_w[o][i][seg]
__global__ void pack_conv_w(const float* __restrict__ in, u16* __restrict__ out) {
  const int n = 2048 * 8192;
  for (int i = blockIdx.x * blockDim.x + threadIdx.x; i < n; i += gridDim.x * blockDim.x) {
    int o   = i >> 13;
    int rem = i & 8191;
    int seg = rem >> 11;
    int ii  = rem & 2047;
    out[i] = f2bf(in[(((size_t)o << 11) + ii) * 4 + seg]);
  }
}

// x_proj_w (33, 2048) -> bf16 padded to (48, 2048), zero rows 33..47
__global__ void pad_xproj_w(const float* __restrict__ in, u16* __restrict__ out) {
  const int n = 48 * 2048;
  for (int i = blockIdx.x * blockDim.x + threadIdx.x; i < n; i += gridDim.x * blockDim.x) {
    int r = i >> 11, c = i & 2047;
    out[i] = (r < 33) ? f2bf(in[r * 2048 + c]) : (u16)0;
  }
}

// ---------------------------------------------------------------------------
// LayerNorm over 1024 features, one token per block, output bf16
// ---------------------------------------------------------------------------
__global__ __launch_bounds__(256) void layernorm_bf16(
    const float* __restrict__ x, const float* __restrict__ w,
    const float* __restrict__ b, u16* __restrict__ out) {
  __shared__ float red[256];
  const int tok = blockIdx.x;
  const int tid = threadIdx.x;
  const float* xr = x + (size_t)tok * 1024;
  float4 v = *(const float4*)(xr + tid * 4);

  float s = v.x + v.y + v.z + v.w;
  red[tid] = s;
  __syncthreads();
  for (int o = 128; o > 0; o >>= 1) { if (tid < o) red[tid] += red[tid + o]; __syncthreads(); }
  const float mu = red[0] * (1.f / 1024.f);
  __syncthreads();

  float dx = v.x - mu, dy = v.y - mu, dz = v.z - mu, dw = v.w - mu;
  red[tid] = dx * dx + dy * dy + dz * dz + dw * dw;
  __syncthreads();
  for (int o = 128; o > 0; o >>= 1) { if (tid < o) red[tid] += red[tid + o]; __syncthreads(); }
  const float rstd = rsqrtf(red[0] * (1.f / 1024.f) + 1e-5f);

  const int c = tid * 4;
  ushort4 o4;
  o4.x = f2bf(dx * rstd * w[c + 0] + b[c + 0]);
  o4.y = f2bf(dy * rstd * w[c + 1] + b[c + 1]);
  o4.z = f2bf(dz * rstd * w[c + 2] + b[c + 2]);
  o4.w = f2bf(dw * rstd * w[c + 3] + b[c + 3]);
  *(ushort4*)(out + (size_t)tok * 1024 + c) = o4;
}

// ---------------------------------------------------------------------------
// Tiled bf16 WMMA GEMM: C(M,N) = A(M,K) * B(N,K)^T , f32 accumulate.
// 128x128 workgroup tile, BK=32, 256 threads (8 waves, each 32x64).
// Double-buffered LDS; tiles streamed with GLOBAL_LOAD_ASYNC_TO_LDS_B128.
// MODE 0: in_proj  -> n<2048: x_in bf16 ; n>=2048: silu(z) f32
// MODE 1: conv     -> A rows shifted per K-segment (causal conv); +bias, silu
//                     -> x_act f32 AND x_act bf16
// MODE 2: x_proj   -> f32 out, row stride 48, only n < nvalid stored
// MODE 3: out_proj -> f32 out = acc + residual
// ---------------------------------------------------------------------------
template <int MODE>
__global__ __launch_bounds__(256) void gemm_bf16_wmma(
    const u16* __restrict__ A, const u16* __restrict__ B,
    int M, int N, int K,
    const float* __restrict__ bias, const float* __restrict__ resid,
    float* __restrict__ outf, u16* __restrict__ outbf, int nvalid) {
  constexpr int BM = 128, BN = 128, BK = 32;
  constexpr int LDAs = BK + 8;   // +16B pad per row, keeps 16B alignment
  constexpr int LDBs = BK + 8;
  __shared__ __align__(16) u16 sA[2][BM * LDAs];
  __shared__ __align__(16) u16 sB[2][BN * LDBs];

  const int tid   = threadIdx.x;
  const int lane  = tid & 31;
  const int wid   = tid >> 5;
  const int wm    = wid & 3;          // 4 wave-rows  (32 rows each)
  const int wn    = wid >> 2;         // 2 wave-cols  (64 cols each)
  const int mlane = lane & 15;
  const int hl    = lane >> 4;

  const int mblk = blockIdx.y * BM;
  const int nblk = blockIdx.x * BN;

  v8f acc[2][4];
  const v8f vzero = {};
#pragma unroll
  for (int i = 0; i < 2; ++i)
#pragma unroll
    for (int j = 0; j < 4; ++j) acc[i][j] = vzero;

  // ---- stage one 128x32 A tile and 128x32 B tile into LDS buffer `buf` ----
  auto stage = [&](int k0, int buf) {
#pragma unroll
    for (int j = 0; j < 2; ++j) {
      const int c   = tid + j * 256;
      const int row = c >> 2;
      const int c8  = (c & 3) * 8;           // bf16 offset within BK

      u16* dstA = &sA[buf][row * LDAs + c8];
      if (MODE == 1) {
        // A is x_in bf16 (4096 rows x 2048); K-segment seg uses rows s+seg-3
        const int seg = k0 >> 11;
        const int ki  = (k0 & 2047) + c8;
        const int gr  = mblk + row;
        const int bb  = gr >> 11;
        const int ss  = (gr & 2047) + seg - 3;
        if (ss >= 0) {
          async_copy_b128(lds_off(dstA),
                          A + (((size_t)(bb << 11) + ss) * 2048 + ki));
        } else {
          const uint4 z = {0u, 0u, 0u, 0u};
          *(uint4*)dstA = z;
        }
      } else {
        async_copy_b128(lds_off(dstA),
                        A + ((size_t)(mblk + row) * K + k0 + c8));
      }

      u16* dstB = &sB[buf][row * LDBs + c8];
      const int nr = nblk + row;
      if (nr < N) {
        async_copy_b128(lds_off(dstB), B + ((size_t)nr * K + k0 + c8));
      } else {
        const uint4 z = {0u, 0u, 0u, 0u};
        *(uint4*)dstB = z;
      }
    }
  };

  const int nIters = K / BK;
  stage(0, 0);

  for (int it = 0; it < nIters; ++it) {
    const int buf = it & 1;
    wait_async0();          // this wave's portion of tile `it` has landed in LDS
    __syncthreads();        // whole tile `it` visible; everyone done reading buf^1

    if (it + 1 < nIters) stage((it + 1) * BK, buf ^ 1);   // prefetch next tile

    // ---- fragments + WMMA on buffer `buf` ----
    v16bf bfr[4];
#pragma unroll
    for (int ni = 0; ni < 4; ++ni) {
      const u16* p = &sB[buf][(wn * 64 + ni * 16 + mlane) * LDBs + hl * 8];
      bfr[ni] = make_frag(*(const uint4*)p, *(const uint4*)(p + 16));
    }
#pragma unroll
    for (int mi = 0; mi < 2; ++mi) {
      const u16* p = &sA[buf][(wm * 32 + mi * 16 + mlane) * LDAs];
      v16bf afr = make_frag(*(const uint4*)(p + hl * 8),
                            *(const uint4*)(p + 16 + hl * 8));
#pragma unroll
      for (int ni = 0; ni < 4; ++ni)
        acc[mi][ni] = __builtin_amdgcn_wmma_f32_16x16x32_bf16(
            false, afr, false, bfr[ni], (short)0, acc[mi][ni], false, false);
    }
  }

  // ---- epilogue ----
#pragma unroll
  for (int mi = 0; mi < 2; ++mi) {
#pragma unroll
    for (int ni = 0; ni < 4; ++ni) {
      const int gn  = nblk + wn * 64 + ni * 16 + mlane;
      const int gm0 = mblk + wm * 32 + mi * 16 + hl * 8;
#pragma unroll
      for (int r = 0; r < 8; ++r) {
        const float v = acc[mi][ni][r];
        const int gm = gm0 + r;
        if (MODE == 0) {
          if (gn < 2048) outbf[(size_t)gm * 2048 + gn] = f2bf(v);
          else           outf [(size_t)gm * 2048 + (gn - 2048)] = siluf(v);
        } else if (MODE == 1) {
          const float s = siluf(v + bias[gn]);
          outf [(size_t)gm * 2048 + gn] = s;
          outbf[(size_t)gm * 2048 + gn] = f2bf(s);
        } else if (MODE == 2) {
          if (gn < nvalid) outf[(size_t)gm * 48 + gn] = v;
        } else {
          outf[(size_t)gm * 1024 + gn] = v + resid[(size_t)gm * 1024 + gn];
        }
      }
    }
  }
}

// ---------------------------------------------------------------------------
// Selective-scan: 4 blocks (2 batches x 2 channel halves) x 1024 threads.
// Each thread owns one channel d with its full 16-state vector in registers.
// x_ssm row layout (stride 48): [delta_s, B0..B15, C0..C15, pad...]
// Output: g = y * silu(z) as bf16 (input to out_proj GEMM).
// ---------------------------------------------------------------------------
__global__ __launch_bounds__(1024) void scan_kernel(
    const float* __restrict__ xssm, const float* __restrict__ u_f32,
    const float* __restrict__ zsilu, const float* __restrict__ dtw,
    const float* __restrict__ dtb, u16* __restrict__ g) {
  const int bb = blockIdx.x >> 1;
  const int d  = ((blockIdx.x & 1) << 10) + threadIdx.x;
  __shared__ float sx[64 * 48];

  float h[16];
#pragma unroll
  for (int j = 0; j < 16; ++j) h[j] = 0.f;
  const float w  = dtw[d];
  const float b0 = dtb[d];

  for (int ch = 0; ch < 2048; ch += 64) {
    __syncthreads();
    for (int i = threadIdx.x; i < 64 * 48; i += 1024)
      sx[i] = xssm[((size_t)(bb * 2048 + ch)) * 48 + i];
    __syncthreads();

    for (int tt = 0; tt < 64; ++tt) {
      const size_t base = ((size_t)(bb * 2048 + ch + tt)) * 2048 + d;
      const float a     = sx[tt * 48] * w + b0;
      const float delta = (a > 20.f) ? a : log1pf(__expf(a));
      const float decay = __expf(-delta);
      const float u     = u_f32[base];
      float y = 0.f;
#pragma unroll
      for (int j = 0; j < 16; ++j) {
        h[j] = h[j] * decay + sx[tt * 48 + 1 + j] * u;
        y += h[j] * sx[tt * 48 + 17 + j];
      }
      g[base] = f2bf(y * zsilu[base]);
    }
  }
}

// ---------------------------------------------------------------------------
// Host-side orchestration
// ---------------------------------------------------------------------------
extern "C" void kernel_launch(void* const* d_in, const int* in_sizes, int n_in,
                              void* d_out, int out_size, void* d_ws, size_t ws_size,
                              hipStream_t stream) {
  (void)in_sizes; (void)n_in; (void)out_size; (void)ws_size;

  const float* x        = (const float*)d_in[0];  // (2, 2048, 1024)
  const float* norm_w   = (const float*)d_in[1];  // (1024,)
  const float* norm_b   = (const float*)d_in[2];  // (1024,)
  const float* in_w     = (const float*)d_in[3];  // (4096, 1024)
  const float* conv_w   = (const float*)d_in[4];  // (2048, 2048, 4)
  const float* conv_b   = (const float*)d_in[5];  // (2048,)
  const float* xproj_w  = (const float*)d_in[6];  // (33, 2048)
  const float* dt_w     = (const float*)d_in[7];  // (2048, 1)
  const float* dt_b     = (const float*)d_in[8];  // (2048,)
  const float* out_w    = (const float*)d_in[9];  // (1024, 2048)
  float*       out      = (float*)d_out;          // (2, 2048, 1024)

  size_t off = 0;
  char* ws = (char*)d_ws;
  auto alloc = [&](size_t bytes) -> void* {
    void* p = ws + off;
    off = (off + bytes + 255) & ~(size_t)255;
    return p;
  };

  u16*   xn_bf     = (u16*)  alloc(4096ull * 1024 * 2);   // normalized input, bf16
  u16*   win_bf    = (u16*)  alloc(4096ull * 1024 * 2);   // in_proj_w bf16
  u16*   xin_bf    = (u16*)  alloc(4096ull * 2048 * 2);   // x_in bf16
  float* zsilu     = (float*)alloc(4096ull * 2048 * 4);   // silu(z) f32
  u16*   convw_bf  = (u16*)  alloc(2048ull * 8192 * 2);   // packed conv weight bf16
  float* xact_f    = (float*)alloc(4096ull * 2048 * 4);   // silu(conv) f32 (scan u)
  u16*   xact_bf   = (u16*)  alloc(4096ull * 2048 * 2);   // silu(conv) bf16 (x_proj A)
  u16*   xprojw_bf = (u16*)  alloc(48ull   * 2048 * 2);   // padded x_proj_w bf16
  float* xssm      = (float*)alloc(4096ull * 48 * 4);     // [dt_s | B | C] stride 48
  u16*   g_bf      = (u16*)  alloc(4096ull * 2048 * 2);   // y * silu(z) bf16
  u16*   outw_bf   = (u16*)  alloc(1024ull * 2048 * 2);   // out_proj_w bf16

  // Weight conversions / packing
  cvt_bf16   <<<2048, 256, 0, stream>>>(in_w,  win_bf,  4096 * 1024);
  pack_conv_w<<<8192, 256, 0, stream>>>(conv_w, convw_bf);
  pad_xproj_w<<<  96, 256, 0, stream>>>(xproj_w, xprojw_bf);
  cvt_bf16   <<<2048, 256, 0, stream>>>(out_w, outw_bf, 1024 * 2048);

  // 1) LayerNorm -> bf16
  layernorm_bf16<<<4096, 256, 0, stream>>>(x, norm_w, norm_b, xn_bf);

  // 2) in_proj: (4096x1024)*(1024x4096) -> x_in bf16 + silu(z) f32
  gemm_bf16_wmma<0><<<dim3(32, 32), 256, 0, stream>>>(
      xn_bf, win_bf, 4096, 4096, 1024, nullptr, nullptr, zsilu, xin_bf, 4096);

  // 3) causal conv as K=8192 GEMM with shifted A segments; +bias, silu
  gemm_bf16_wmma<1><<<dim3(16, 32), 256, 0, stream>>>(
      xin_bf, convw_bf, 4096, 2048, 8192, conv_b, nullptr, xact_f, xact_bf, 2048);

  // 4) x_proj: (4096x2048)*(2048x48pad) -> x_ssm f32 (33 valid cols, stride 48)
  gemm_bf16_wmma<2><<<dim3(1, 32), 256, 0, stream>>>(
      xact_bf, xprojw_bf, 4096, 48, 2048, nullptr, nullptr, xssm, nullptr, 33);

  // 5) selective scan (register-resident state) -> g = y * silu(z), bf16
  scan_kernel<<<4, 1024, 0, stream>>>(xssm, xact_f, zsilu, dt_w, dt_b, g_bf);

  // 6) out_proj + residual -> d_out f32
  gemm_bf16_wmma<3><<<dim3(8, 32), 256, 0, stream>>>(
      g_bf, outw_bf, 4096, 1024, 2048, nullptr, x, out, nullptr, 1024);
}
